// Speech_MSA_41669772706157
// MI455X (gfx1250) — matrix operand order repaired and checked by reading, not compiled, not run on hardware
//
#include <hip/hip_runtime.h>

// ---- problem constants (from reference) ----
#define B_ 8
#define T_ 4096
#define E_ 1024
#define H_ 16
#define L_ 64
#define HD_ 64
#define NBLK_ (T_ / L_)        // 64
#define E3_ (3 * E_)           // 3072
#define M_ (B_ * T_)           // 32768

typedef __attribute__((ext_vector_type(16))) __bf16 v16bf;
typedef __attribute__((ext_vector_type(8)))  float  v8f;
typedef unsigned short bf16u;  // raw bf16 bits in memory

__device__ __forceinline__ bf16u f2bf(float f) {
    union { float f; unsigned u; } x; x.f = f;
    unsigned r = x.u + 0x7FFFu + ((x.u >> 16) & 1u);   // round-to-nearest-even
    return (bf16u)(r >> 16);
}

union Frag { uint4 u[2]; v16bf v; };

// ---- CDNA5 async global->LDS copy (ASYNCcnt-tracked), via inline asm ----
// vdst = LDS byte address (low 32 bits of generic shared pointer),
// vaddr = 64-bit global address. 16 bytes per lane.
__device__ __forceinline__ void async_ld_b128(unsigned lds_off, const void* gptr) {
    unsigned long long ga = (unsigned long long)(size_t)gptr;
    asm volatile("global_load_async_to_lds_b128 %0, %1, off"
                 :: "v"(lds_off), "v"(ga) : "memory");
}
template <int N>
__device__ __forceinline__ void wait_async() {
    asm volatile("s_wait_asynccnt %0" :: "i"(N) : "memory");
}

// ---- ds_swizzle-based butterfly over 16-lane halves (xor 1,2,4,8) ----
template <int IMM>
__device__ __forceinline__ float swz(float x) {
    return __int_as_float(__builtin_amdgcn_ds_swizzle(__float_as_int(x), IMM));
}
__device__ __forceinline__ float red_max16(float v) {
    v = fmaxf(v, swz<0x041F>(v));  // xor 1
    v = fmaxf(v, swz<0x081F>(v));  // xor 2
    v = fmaxf(v, swz<0x101F>(v));  // xor 4
    v = fmaxf(v, swz<0x201F>(v));  // xor 8
    return v;
}
__device__ __forceinline__ float red_sum16(float v) {
    v += swz<0x041F>(v);
    v += swz<0x081F>(v);
    v += swz<0x101F>(v);
    v += swz<0x201F>(v);
    return v;
}

// ---------------------------------------------------------------------------
// f32 -> bf16 convert (grid-stride)
// ---------------------------------------------------------------------------
__global__ void f32_to_bf16_kernel(const float* __restrict__ in,
                                   bf16u* __restrict__ out, size_t n) {
    size_t i = (size_t)blockIdx.x * blockDim.x + threadIdx.x;
    size_t stride = (size_t)gridDim.x * blockDim.x;
    for (; i < n; i += stride) out[i] = f2bf(in[i]);
}

// ---------------------------------------------------------------------------
// C(MxN) = A(MxK,bf16,row-major) * Wt(NxK,bf16,row-major)^T + bias
// 128x128x32 double-buffered LDS tile staged with async global->LDS loads,
// 256 threads = 8 waves (4 along M x 2 along N), 2x4 WMMA tiles per wave.
// QSCALE: multiply (acc+bias) by qscale for columns n < qcols (q pre-scaling).
// ---------------------------------------------------------------------------
template <bool OUT_F32, bool QSCALE>
__global__ __launch_bounds__(256) void gemm_bf16_kernel(
    const bf16u* __restrict__ A, const bf16u* __restrict__ Wt,
    const float* __restrict__ bias, void* __restrict__ Cout,
    int M, int N, int K, float qscale, int qcols) {
    constexpr int BM = 128, BN = 128, BK = 32, LDP = BK + 8;
    __shared__ __align__(16) bf16u As[2][BM][LDP];
    __shared__ __align__(16) bf16u Bs[2][BN][LDP];

    const int tid  = threadIdx.x;
    const int lane = tid & 31;
    const int wv   = tid >> 5;
    const int wM   = wv & 3;      // wave row (0..3) -> 32 rows each
    const int wN   = wv >> 2;     // wave col (0..1) -> 64 cols each
    const int hg   = lane >> 4;   // half-group 0/1
    const int l16  = lane & 15;

    const int mBlk = blockIdx.y * BM;
    const int nBlk = blockIdx.x * BN;

    v8f acc[2][4] = {};

    // async-stage one 128x32 A tile + 128x32 B tile (4 b128 per thread)
    auto stage = [&](int sbuf, int k0) {
#pragma unroll
        for (int it = 0; it < 2; ++it) {
            int idx = tid + it * 256;
            int r = idx >> 2, c = (idx & 3) * 8;
            async_ld_b128((unsigned)(size_t)&As[sbuf][r][c],
                          &A[(size_t)(mBlk + r) * K + k0 + c]);
            async_ld_b128((unsigned)(size_t)&Bs[sbuf][r][c],
                          &Wt[(size_t)(nBlk + r) * K + k0 + c]);
        }
    };

    const int nk = K / BK;
    int buf = 0;
    stage(0, 0);
    for (int kt = 0; kt < nk; ++kt) {
        if (kt + 1 < nk) {            // prefetch next tile, leave it in flight
            stage(buf ^ 1, (kt + 1) * BK);
            wait_async<4>();          // current tile's 4 loads complete (in-order)
        } else {
            wait_async<0>();
        }
        __syncthreads();

        Frag a[2], b[4];
#pragma unroll
        for (int mi = 0; mi < 2; ++mi) {
            int m = wM * 32 + mi * 16 + l16;
            a[mi].u[0] = *reinterpret_cast<const uint4*>(&As[buf][m][hg * 8]);
            a[mi].u[1] = *reinterpret_cast<const uint4*>(&As[buf][m][16 + hg * 8]);
        }
#pragma unroll
        for (int ni = 0; ni < 4; ++ni) {
            int n = wN * 64 + ni * 16 + l16;
            b[ni].u[0] = *reinterpret_cast<const uint4*>(&Bs[buf][n][hg * 8]);
            b[ni].u[1] = *reinterpret_cast<const uint4*>(&Bs[buf][n][16 + hg * 8]);
        }
#pragma unroll
        for (int mi = 0; mi < 2; ++mi)
#pragma unroll
            for (int ni = 0; ni < 4; ++ni)
                acc[mi][ni] = __builtin_amdgcn_wmma_f32_16x16x32_bf16(
                    false, a[mi].v, false, b[ni].v, (short)0, acc[mi][ni],
                    false, false);
        __syncthreads();              // all reads of `buf` done before restaging
        buf ^= 1;
    }

    // epilogue: C element (m,n): row = base + r + 8*hg, col = base + l16
#pragma unroll
    for (int mi = 0; mi < 2; ++mi) {
#pragma unroll
        for (int ni = 0; ni < 4; ++ni) {
            int n = nBlk + wN * 64 + ni * 16 + l16;
            float bv = bias[n];
            float sc = (QSCALE && (n < qcols)) ? qscale : 1.0f;
#pragma unroll
            for (int r = 0; r < 8; ++r) {
                int m = mBlk + wM * 32 + mi * 16 + r + hg * 8;
                float v = (acc[mi][ni][r] + bv) * sc;
                if (OUT_F32)
                    reinterpret_cast<float*>(Cout)[(size_t)m * N + n] = v;
                else
                    reinterpret_cast<bf16u*>(Cout)[(size_t)m * N + n] = f2bf(v);
            }
        }
    }
}

// ---------------------------------------------------------------------------
// Block-local attention: one workgroup per (b, h, nblk); 128 threads = 4 waves.
// Each wave owns 16 query rows: S = q·k^T (WMMA), register softmax via
// ds_swizzle butterflies over 16-lane halves, P->LDS (bf16, A-frag layout),
// O = P·v (WMMA). qkv: (B,T,3E) bf16 with q pre-scaled. o: (B,T,E) bf16.
// ---------------------------------------------------------------------------
__global__ __launch_bounds__(128) void attn_kernel(
    const bf16u* __restrict__ qkv, bf16u* __restrict__ o) {
    __shared__ __align__(16) bf16u Ks[64][72];        // K block, row-major
    __shared__ __align__(16) bf16u Vs[64][72];        // V block, TRANSPOSED: Vs[d][token]
    __shared__ __align__(16) bf16u Ps[4][16][72];     // per-wave softmax probs

    const int blk  = blockIdx.x;
    const int nblk = blk % NBLK_;
    const int h    = (blk / NBLK_) % H_;
    const int b    = blk / (NBLK_ * H_);

    const int tid  = threadIdx.x;
    const int lane = tid & 31;
    const int wv   = tid >> 5;
    const int hg   = lane >> 4;
    const int l16  = lane & 15;

    const size_t tokBase = ((size_t)b * T_ + (size_t)nblk * L_);

    // ---- stage K (row-major) and V (transposed) into LDS ----
    {
        int r = tid >> 1, c0 = (tid & 1) * 32;  // each thread: half a row (32 elems)
        const bf16u* krow = qkv + (tokBase + r) * E3_ + E_ + h * HD_ + c0;
        const bf16u* vrow = qkv + (tokBase + r) * E3_ + 2 * E_ + h * HD_ + c0;
#pragma unroll
        for (int i = 0; i < 4; ++i)
            *reinterpret_cast<uint4*>(&Ks[r][c0 + i * 8]) =
                *reinterpret_cast<const uint4*>(krow + i * 8);
        union { uint4 u[4]; bf16u s[32]; } vt;
#pragma unroll
        for (int i = 0; i < 4; ++i)
            vt.u[i] = *reinterpret_cast<const uint4*>(vrow + i * 8);
#pragma unroll
        for (int i = 0; i < 32; ++i) Vs[c0 + i][r] = vt.s[i];  // transpose store
    }
    __syncthreads();

    // ---- load q A-fragments straight from global (rows = this wave's 16) ----
    const bf16u* qr = qkv + (tokBase + wv * 16 + l16) * E3_ + h * HD_;
    Frag aq[2];
#pragma unroll
    for (int kk = 0; kk < 2; ++kk) {
        aq[kk].u[0] = *reinterpret_cast<const uint4*>(qr + kk * 32 + hg * 8);
        aq[kk].u[1] = *reinterpret_cast<const uint4*>(qr + kk * 32 + 16 + hg * 8);
    }

    // ---- S = q · k^T : 4 n-tiles x 2 k-steps ----
    v8f s[4] = {};
#pragma unroll
    for (int ni = 0; ni < 4; ++ni) {
#pragma unroll
        for (int kk = 0; kk < 2; ++kk) {
            Frag bk;
            int n = ni * 16 + l16;
            bk.u[0] = *reinterpret_cast<const uint4*>(&Ks[n][kk * 32 + hg * 8]);
            bk.u[1] = *reinterpret_cast<const uint4*>(&Ks[n][kk * 32 + 16 + hg * 8]);
            s[ni] = __builtin_amdgcn_wmma_f32_16x16x32_bf16(
                false, aq[kk].v, false, bk.v, (short)0, s[ni], false, false);
        }
    }

    // ---- row softmax: row's 64 values live in (16 lanes of this half) x 4 tiles ----
#pragma unroll
    for (int r = 0; r < 8; ++r) {
        float mx = fmaxf(fmaxf(s[0][r], s[1][r]), fmaxf(s[2][r], s[3][r]));
        mx = red_max16(mx);
        float sum = 0.f;
#pragma unroll
        for (int ni = 0; ni < 4; ++ni) {
            float e = __expf(s[ni][r] - mx);
            s[ni][r] = e;
            sum += e;
        }
        sum = red_sum16(sum);
        float inv = 1.0f / sum;
#pragma unroll
        for (int ni = 0; ni < 4; ++ni) s[ni][r] *= inv;
    }

    // ---- C-layout -> A-layout through wave-private LDS (same-wave DS is in-order) ----
#pragma unroll
    for (int ni = 0; ni < 4; ++ni)
#pragma unroll
        for (int r = 0; r < 8; ++r)
            Ps[wv][r + hg * 8][ni * 16 + l16] = f2bf(s[ni][r]);

    Frag ap[2];
#pragma unroll
    for (int kk = 0; kk < 2; ++kk) {
        ap[kk].u[0] = *reinterpret_cast<const uint4*>(&Ps[wv][l16][kk * 32 + hg * 8]);
        ap[kk].u[1] = *reinterpret_cast<const uint4*>(&Ps[wv][l16][kk * 32 + 16 + hg * 8]);
    }

    // ---- O = P · v : B frags from transposed Vs (contiguous in token dim) ----
    v8f oa[4] = {};
#pragma unroll
    for (int dj = 0; dj < 4; ++dj) {
#pragma unroll
        for (int kk = 0; kk < 2; ++kk) {
            Frag bv;
            int d = dj * 16 + l16;
            bv.u[0] = *reinterpret_cast<const uint4*>(&Vs[d][kk * 32 + hg * 8]);
            bv.u[1] = *reinterpret_cast<const uint4*>(&Vs[d][kk * 32 + 16 + hg * 8]);
            oa[dj] = __builtin_amdgcn_wmma_f32_16x16x32_bf16(
                false, ap[kk].v, false, bv.v, (short)0, oa[dj], false, false);
        }
    }

    // ---- store O block to (B,T,E) bf16 layout for the output GEMM ----
#pragma unroll
    for (int dj = 0; dj < 4; ++dj)
#pragma unroll
        for (int r = 0; r < 8; ++r) {
            int trow = wv * 16 + r + hg * 8;
            int dcol = dj * 16 + l16;
            o[(tokBase + trow) * E_ + h * HD_ + dcol] = f2bf(oa[dj][r]);
        }
}

// ---------------------------------------------------------------------------
// launch
// ---------------------------------------------------------------------------
extern "C" void kernel_launch(void* const* d_in, const int* in_sizes, int n_in,
                              void* d_out, int out_size, void* d_ws, size_t ws_size,
                              hipStream_t stream) {
    const float* x     = (const float*)d_in[0];  // (B,T,E)
    const float* W_qkv = (const float*)d_in[1];  // (3E,E)
    const float* b_qkv = (const float*)d_in[2];  // (3E)
    const float* W_out = (const float*)d_in[3];  // (E,E)
    const float* b_out = (const float*)d_in[4];  // (E)
    float* out = (float*)d_out;

    const size_t n_x    = (size_t)M_ * E_;        // 33,554,432
    const size_t n_wqkv = (size_t)E3_ * E_;       //  3,145,728
    const size_t n_wout = (size_t)E_ * E_;        //  1,048,576
    const size_t n_qkv  = (size_t)M_ * E3_;       // 100,663,296

    bf16u* ws    = (bf16u*)d_ws;                  // ~344 MB total bf16 workspace
    bf16u* xbf   = ws;
    bf16u* wqkvb = xbf + n_x;
    bf16u* woutb = wqkvb + n_wqkv;
    bf16u* qkvb  = woutb + n_wout;
    bf16u* obf   = qkvb + n_qkv;

    f32_to_bf16_kernel<<<4096, 256, 0, stream>>>(x, xbf, n_x);
    f32_to_bf16_kernel<<<1024, 256, 0, stream>>>(W_qkv, wqkvb, n_wqkv);
    f32_to_bf16_kernel<<<512,  256, 0, stream>>>(W_out, woutb, n_wout);

    // qkv = x @ W_qkv^T + b_qkv, then q *= HD^-0.5 (first E columns)
    gemm_bf16_kernel<false, true><<<dim3(E3_ / 128, M_ / 128), 256, 0, stream>>>(
        xbf, wqkvb, b_qkv, qkvb, M_, E3_, E_, 0.125f, E_);

    attn_kernel<<<B_ * H_ * NBLK_, 128, 0, stream>>>(qkvb, obf);

    // out = o @ W_out^T + b_out (f32 output)
    gemm_bf16_kernel<true, false><<<dim3(E_ / 128, M_ / 128), 256, 0, stream>>>(
        obf, woutb, b_out, out, M_, E_, E_, 1.0f, 0);
}
